// DeepseekV2MoE_64828236366676
// MI455X (gfx1250) — compile-verified, matrix operand order checked by reading, not run on hardware
//
#include <hip/hip_runtime.h>
#include <hip/hip_bf16.h>
#include <math.h>
#include <stdint.h>

typedef __attribute__((ext_vector_type(16))) __bf16 v16bf;
typedef __attribute__((ext_vector_type(8)))  float  v8f;
typedef __attribute__((ext_vector_type(4)))  unsigned v4u;
typedef __attribute__((ext_vector_type(8)))  int    v8i;
typedef __attribute__((ext_vector_type(4)))  int    v4i;

constexpr int Hd  = 2048;   // hidden
constexpr int Ed  = 64;     // experts
constexpr int Md  = 1408;   // expert intermediate
constexpr int MSd = 2816;   // shared intermediate
constexpr int Td  = 4096;   // tokens
constexpr int Kd  = 6;      // top-k
constexpr int Cd  = 768;    // capacity
constexpr int NGd = 8;      // groups
constexpr int TGd = 3;      // top groups

#if __has_builtin(__builtin_amdgcn_tensor_load_to_lds)
#define USE_TDM 1
#endif

static __device__ __forceinline__ __bf16 f2bf(float f) {
  unsigned u = __builtin_bit_cast(unsigned, f);
  unsigned r = u + 0x7FFFu + ((u >> 16) & 1u);   // round-to-nearest-even
  unsigned short h = (unsigned short)(r >> 16);
  return __builtin_bit_cast(__bf16, h);
}

#ifdef USE_TDM
// TDM: DMA a [64 rows x 32 cols] bf16 tile (row stride K elements) from global
// into LDS at lds_byte_off. 2D descriptor; groups 2/3 zero (<=2D tensor).
static __device__ __forceinline__ void tdm_load_tile_64x32(
    const __bf16* gsrc, unsigned lds_byte_off, int K) {
  unsigned long long ga = (unsigned long long)(uintptr_t)gsrc;
  v4u g0;
  g0[0] = 1u;                                           // count=1, user desc
  g0[1] = lds_byte_off;                                 // lds_addr
  g0[2] = (unsigned)(ga & 0xFFFFFFFFu);                 // global_addr[31:0]
  g0[3] = (unsigned)((ga >> 32) & 0x1FFFFFFu) | (2u << 30); // addr[56:32] | type=2
  v8i g1;
  g1[0] = (int)(1u << 16);                              // data_size=2B
  g1[1] = (int)(32u << 16);                             // tensor_dim0=32 (tile-local)
  g1[2] = (int)(64u << 16);                             // tensor_dim1=64
  g1[3] = (int)(32u << 16);                             // tile_dim0=32
  g1[4] = 64;                                           // tile_dim1=64, tile_dim2=0
  g1[5] = K;                                            // tensor_dim0_stride=K
  g1[6] = 0;
  g1[7] = 0;
  v4i g2 = {0, 0, 0, 0};
  v4i g3 = {0, 0, 0, 0};
#if defined(__clang_major__) && (__clang_major__ >= 23)
  v8i g4 = {0, 0, 0, 0, 0, 0, 0, 0};
  __builtin_amdgcn_tensor_load_to_lds(g0, g1, g2, g3, g4, 0);
#else
  __builtin_amdgcn_tensor_load_to_lds(g0, g1, g2, g3, 0);
#endif
}
#endif

// ---------------- elementwise f32 -> bf16 ----------------
__global__ void cvt_f32_bf16(const float* __restrict__ in, __bf16* __restrict__ out, size_t n) {
  size_t i = (size_t)blockIdx.x * blockDim.x + threadIdx.x;
  size_t stride = (size_t)gridDim.x * blockDim.x;
  for (; i < n; i += stride) out[i] = f2bf(in[i]);
}

// ---------------- gate: softmax + group-limited greedy top-k ----------------
__global__ __launch_bounds__(128)
void gate_kernel(const float* __restrict__ x, const float* __restrict__ gw,
                 float* __restrict__ topk_w, int* __restrict__ topk_i) {
  __shared__ float xs[4][Hd];
  __shared__ float sl[4][Ed];
  const int wave = threadIdx.x >> 5;
  const int lane = threadIdx.x & 31;
  const int t = blockIdx.x * 4 + wave;
  const float* xr = x + (size_t)t * Hd;
  for (int h = lane; h < Hd; h += 32) xs[wave][h] = xr[h];
  __syncthreads();
  for (int e = 0; e < Ed; ++e) {
    const float* wr = gw + (size_t)e * Hd;
    float p = 0.f;
    for (int h = lane; h < Hd; h += 32) p += xs[wave][h] * wr[h];
    for (int off = 16; off > 0; off >>= 1) p += __shfl_down(p, off, 32);
    if (lane == 0) sl[wave][e] = p;
  }
  if (lane == 0) {
    float* L = sl[wave];
    float mx = L[0];
    for (int e = 1; e < Ed; ++e) mx = fmaxf(mx, L[e]);
    float s = 0.f;
    for (int e = 0; e < Ed; ++e) { L[e] = __expf(L[e] - mx); s += L[e]; }
    float inv = 1.f / s;
    for (int e = 0; e < Ed; ++e) L[e] *= inv;
    float gs[NGd];
    for (int g = 0; g < NGd; ++g) {
      float m = L[g * 8];
      for (int j = 1; j < 8; ++j) m = fmaxf(m, L[g * 8 + j]);
      gs[g] = m;
    }
    unsigned gmask = 0;
    for (int r = 0; r < TGd; ++r) {
      int bi = 0; float bv = -1.f;
      for (int g = 0; g < NGd; ++g)
        if (!((gmask >> g) & 1u) && gs[g] > bv) { bv = gs[g]; bi = g; }
      gmask |= 1u << bi;
    }
    float ms[Ed];
    for (int e = 0; e < Ed; ++e)
      ms[e] = ((gmask >> (e >> 3)) & 1u) ? L[e] : 0.f;
    for (int k = 0; k < Kd; ++k) {
      int bi = 0; float bv = -1.f;
      for (int e = 0; e < Ed; ++e)
        if (ms[e] > bv) { bv = ms[e]; bi = e; }
      topk_w[(size_t)t * Kd + k] = bv;   // RSF == 1.0
      topk_i[(size_t)t * Kd + k] = bi;
      ms[bi] = -1.f;
    }
  }
}

// ---------------- deterministic token-order rank -> slot ----------------
__global__ void slot_kernel(const int* __restrict__ topk_i, int* __restrict__ slot) {
  int e = threadIdx.x;   // one thread per expert
  int cnt = 0;
  for (int i = 0; i < Td * Kd; ++i) {
    if (topk_i[i] == e) {
      int p = cnt < Cd - 1 ? cnt : Cd - 1;
      slot[i] = e * Cd + p;
      ++cnt;
    }
  }
}

// ---------------- scatter tokens into [E,C,H] bf16 buffers ----------------
__global__ __launch_bounds__(256)
void dispatch_kernel(const float* __restrict__ x, const int* __restrict__ slot,
                     __bf16* __restrict__ buf) {
  const int i = blockIdx.x;        // 0 .. T*K-1
  const int t = i / Kd;
  const int s = slot[i];
  const float* src = x + (size_t)t * Hd;
  __bf16* dst = buf + (size_t)s * Hd;
  for (int h = threadIdx.x; h < Hd; h += 256) dst[h] = f2bf(src[h]);
}

// ---------------- fused gate/up GEMM + SwiGLU (WMMA bf16) ----------------
// A: [rows,K] bf16 row-major.  Bg,Bu: [N,K] bf16 row-major ("NT" GEMM).
// Act[r,n] = silu(A.Bg^T) * (A.Bu^T), bf16.
// Block: 256 thr = 8 waves; block tile 64 rows x 128 cols; wave = 16x16 x 4 n-tiles.
// A tile staged via TDM (double-buffered) into LDS.
__global__ __launch_bounds__(256)
void gemm_swiglu_wmma(const __bf16* __restrict__ Abase,
                      const __bf16* __restrict__ Bg_base,
                      const __bf16* __restrict__ Bu_base,
                      __bf16* __restrict__ Actbase,
                      int K, int N,
                      size_t sA, size_t sB, size_t sAct) {
  const __bf16* A  = Abase   + (size_t)blockIdx.z * sA;
  const __bf16* Bg = Bg_base + (size_t)blockIdx.z * sB;
  const __bf16* Bu = Bu_base + (size_t)blockIdx.z * sB;
  __bf16*      Act = Actbase + (size_t)blockIdx.z * sAct;

  __shared__ alignas(32) __bf16 As[2][64 * 32];

  const int tid  = threadIdx.x;
  const int lane = tid & 31;
  const int wave = tid >> 5;
  const int wr   = wave & 3;          // wave row (16 rows each)
  const int wc   = wave >> 2;         // wave col (4 n-tiles of 16 each)
  const int c0   = blockIdx.y * 64;
  const int n0   = blockIdx.x * 128;
  const int lrow = lane & 15;
  const int kseg = (lane >> 4) << 4;  // 0 or 16

  v8f accg[4] = {v8f{}, v8f{}, v8f{}, v8f{}};
  v8f accu[4] = {v8f{}, v8f{}, v8f{}, v8f{}};

#ifdef USE_TDM
  const unsigned lds0 = (unsigned)(uintptr_t)&As[0][0];
  const unsigned lds1 = (unsigned)(uintptr_t)&As[1][0];
  if (wave == 0 && lane == 0)
    tdm_load_tile_64x32(A + (size_t)c0 * K, lds0, K);
#endif
  int cur = 0;

  for (int kk = 0; kk < K; kk += 32) {
#ifdef USE_TDM
    if (wave == 0 && lane == 0) {
      __builtin_amdgcn_s_wait_tensorcnt(0);   // current tile resident
    }
    __syncthreads();
    if (wave == 0 && lane == 0 && (kk + 32) < K) {
      // kick next tile DMA; overlaps with this iteration's WMMAs
      tdm_load_tile_64x32(A + (size_t)c0 * K + (kk + 32), cur ? lds0 : lds1, K);
    }
#else
    {
      const int r = tid >> 2;
      const int c = (tid & 3) << 3;
      *(uint4*)(&As[cur][r * 32 + c]) =
          *(const uint4*)(A + (size_t)(c0 + r) * K + kk + c);
    }
    __syncthreads();
#endif

    const v16bf a = *(const v16bf*)(&As[cur][(wr * 16 + lrow) * 32 + kseg]);

#pragma unroll
    for (int nt = 0; nt < 4; ++nt) {
      const int m = n0 + (wc * 4 + nt) * 16 + lrow;
      const v16bf fg = *(const v16bf*)(Bg + (size_t)m * K + kk + kseg);
      const v16bf fu = *(const v16bf*)(Bu + (size_t)m * K + kk + kseg);
      __builtin_prefetch(Bg + (size_t)m * K + kk + 64, 0, 1);
      __builtin_prefetch(Bu + (size_t)m * K + kk + 64, 0, 1);
      accg[nt] = __builtin_amdgcn_wmma_f32_16x16x32_bf16(
          false, a, false, fg, (short)0, accg[nt], false, false);
      accu[nt] = __builtin_amdgcn_wmma_f32_16x16x32_bf16(
          false, a, false, fu, (short)0, accu[nt], false, false);
    }
    __syncthreads();
    cur ^= 1;
  }

  // epilogue: silu(g)*u -> bf16.  C/D layout: lanes 0-15: M=r, lanes 16-31: M=8+r
  const int rofs = (lane >> 4) << 3;
#pragma unroll
  for (int nt = 0; nt < 4; ++nt) {
    const int mcol = n0 + (wc * 4 + nt) * 16 + lrow;
#pragma unroll
    for (int r = 0; r < 8; ++r) {
      const int crow = c0 + wr * 16 + r + rofs;
      const float g = accg[nt][r];
      const float u = accu[nt][r];
      const float s = g / (1.0f + __expf(-g));
      Act[(size_t)crow * N + mcol] = f2bf(s * u);
    }
  }
}

// ---------------- down-proj GEMM (WMMA bf16) ----------------
// A: [rows,K] bf16; B: [N,K] bf16; Out[r,n] = A.B^T.
// store_bf16 != 0 -> bf16 output, else f32 output.
__global__ __launch_bounds__(256)
void gemm_down_wmma(const __bf16* __restrict__ Abase,
                    const __bf16* __restrict__ Bbase,
                    void* __restrict__ Outbase,
                    int K, int N, int store_bf16,
                    size_t sA, size_t sB, size_t sOut) {
  const __bf16* A = Abase + (size_t)blockIdx.z * sA;
  const __bf16* B = Bbase + (size_t)blockIdx.z * sB;
  const size_t outOfs = (size_t)blockIdx.z * sOut;

  __shared__ alignas(32) __bf16 As[2][64 * 32];

  const int tid  = threadIdx.x;
  const int lane = tid & 31;
  const int wave = tid >> 5;
  const int wr   = wave & 3;
  const int wc   = wave >> 2;
  const int c0   = blockIdx.y * 64;
  const int n0   = blockIdx.x * 128;
  const int lrow = lane & 15;
  const int kseg = (lane >> 4) << 4;

  v8f acc[4] = {v8f{}, v8f{}, v8f{}, v8f{}};

#ifdef USE_TDM
  const unsigned lds0 = (unsigned)(uintptr_t)&As[0][0];
  const unsigned lds1 = (unsigned)(uintptr_t)&As[1][0];
  if (wave == 0 && lane == 0)
    tdm_load_tile_64x32(A + (size_t)c0 * K, lds0, K);
#endif
  int cur = 0;

  for (int kk = 0; kk < K; kk += 32) {
#ifdef USE_TDM
    if (wave == 0 && lane == 0) {
      __builtin_amdgcn_s_wait_tensorcnt(0);
    }
    __syncthreads();
    if (wave == 0 && lane == 0 && (kk + 32) < K) {
      tdm_load_tile_64x32(A + (size_t)c0 * K + (kk + 32), cur ? lds0 : lds1, K);
    }
#else
    {
      const int r = tid >> 2;
      const int c = (tid & 3) << 3;
      *(uint4*)(&As[cur][r * 32 + c]) =
          *(const uint4*)(A + (size_t)(c0 + r) * K + kk + c);
    }
    __syncthreads();
#endif

    const v16bf a = *(const v16bf*)(&As[cur][(wr * 16 + lrow) * 32 + kseg]);

#pragma unroll
    for (int nt = 0; nt < 4; ++nt) {
      const int m = n0 + (wc * 4 + nt) * 16 + lrow;
      const v16bf fb = *(const v16bf*)(B + (size_t)m * K + kk + kseg);
      __builtin_prefetch(B + (size_t)m * K + kk + 64, 0, 1);
      acc[nt] = __builtin_amdgcn_wmma_f32_16x16x32_bf16(
          false, a, false, fb, (short)0, acc[nt], false, false);
    }
    __syncthreads();
    cur ^= 1;
  }

  const int rofs = (lane >> 4) << 3;
#pragma unroll
  for (int nt = 0; nt < 4; ++nt) {
    const int mcol = n0 + (wc * 4 + nt) * 16 + lrow;
#pragma unroll
    for (int r = 0; r < 8; ++r) {
      const int crow = c0 + wr * 16 + r + rofs;
      const size_t idx = outOfs + (size_t)crow * N + mcol;
      if (store_bf16) ((__bf16*)Outbase)[idx] = f2bf(acc[nt][r]);
      else            ((float*)Outbase)[idx]  = acc[nt][r];
    }
  }
}

// ---------------- combine: y += sum_k w_k * out[slot_k] ----------------
__global__ __launch_bounds__(256)
void combine_kernel(const __bf16* __restrict__ obuf,
                    const float* __restrict__ topk_w,
                    const int* __restrict__ slot,
                    float* __restrict__ y) {
  const int t = blockIdx.x;
  float w[Kd]; int s[Kd];
#pragma unroll
  for (int k = 0; k < Kd; ++k) {
    w[k] = topk_w[(size_t)t * Kd + k];
    s[k] = slot[(size_t)t * Kd + k];
  }
  for (int h = threadIdx.x; h < Hd; h += 256) {
    float acc = y[(size_t)t * Hd + h];  // shared-expert part already stored
#pragma unroll
    for (int k = 0; k < Kd; ++k)
      acc += w[k] * (float)obuf[(size_t)s[k] * Hd + h];
    y[(size_t)t * Hd + h] = acc;
  }
}

extern "C" void kernel_launch(void* const* d_in, const int* in_sizes, int n_in,
                              void* d_out, int out_size, void* d_ws, size_t ws_size,
                              hipStream_t stream) {
  (void)in_sizes; (void)n_in; (void)out_size; (void)ws_size;
  const float* x  = (const float*)d_in[0];
  const float* gw = (const float*)d_in[1];
  const float* Wg = (const float*)d_in[2];
  const float* Wu = (const float*)d_in[3];
  const float* Wd = (const float*)d_in[4];
  const float* Sg = (const float*)d_in[5];
  const float* Su = (const float*)d_in[6];
  const float* Sd = (const float*)d_in[7];
  float* out = (float*)d_out;

  char* p = (char*)d_ws;
  auto alloc = [&](size_t bytes) {
    char* r = p;
    p += (bytes + 255) & ~(size_t)255;
    return r;
  };
  float*  topk_w = (float*) alloc((size_t)Td * Kd * 4);
  int*    topk_i = (int*)   alloc((size_t)Td * Kd * 4);
  int*    slot   = (int*)   alloc((size_t)Td * Kd * 4);
  __bf16* xbf    = (__bf16*)alloc((size_t)Td * Hd * 2);
  __bf16* buf    = (__bf16*)alloc((size_t)Ed * Cd * Hd * 2);
  __bf16* act    = (__bf16*)alloc((size_t)Ed * Cd * Md * 2);
  __bf16* act_s  = (__bf16*)alloc((size_t)Td * MSd * 2);
  __bf16* obuf   = (__bf16*)alloc((size_t)Ed * Cd * Hd * 2);
  __bf16* Wgbf   = (__bf16*)alloc((size_t)Ed * Md * Hd * 2);
  __bf16* Wubf   = (__bf16*)alloc((size_t)Ed * Md * Hd * 2);
  __bf16* Wdbf   = (__bf16*)alloc((size_t)Ed * Hd * Md * 2);
  __bf16* Sgbf   = (__bf16*)alloc((size_t)MSd * Hd * 2);
  __bf16* Subf   = (__bf16*)alloc((size_t)MSd * Hd * 2);
  __bf16* Sdbf   = (__bf16*)alloc((size_t)Hd * MSd * 2);

  // 0) one-time streaming conversion of weights + activations to bf16
  cvt_f32_bf16<<<16384, 256, 0, stream>>>(Wg, Wgbf, (size_t)Ed * Md * Hd);
  cvt_f32_bf16<<<16384, 256, 0, stream>>>(Wu, Wubf, (size_t)Ed * Md * Hd);
  cvt_f32_bf16<<<16384, 256, 0, stream>>>(Wd, Wdbf, (size_t)Ed * Hd * Md);
  cvt_f32_bf16<<<4096,  256, 0, stream>>>(Sg, Sgbf, (size_t)MSd * Hd);
  cvt_f32_bf16<<<4096,  256, 0, stream>>>(Su, Subf, (size_t)MSd * Hd);
  cvt_f32_bf16<<<4096,  256, 0, stream>>>(Sd, Sdbf, (size_t)Hd * MSd);
  cvt_f32_bf16<<<4096,  256, 0, stream>>>(x,  xbf,  (size_t)Td * Hd);
  // 1) router
  gate_kernel<<<Td / 4, 128, 0, stream>>>(x, gw, topk_w, topk_i);
  // 2) deterministic rank/slot (token order, capacity-clamped)
  slot_kernel<<<1, 64, 0, stream>>>(topk_i, slot);
  // 3) zero + scatter dispatch buffers
  hipMemsetAsync(buf, 0, (size_t)Ed * Cd * Hd * 2, stream);
  dispatch_kernel<<<Td * Kd, 256, 0, stream>>>(x, slot, buf);
  // 4) expert gate/up + SwiGLU
  gemm_swiglu_wmma<<<dim3(Md / 128, Cd / 64, Ed), 256, 0, stream>>>(
      buf, Wgbf, Wubf, act, Hd, Md,
      (size_t)Cd * Hd, (size_t)Md * Hd, (size_t)Cd * Md);
  // 5) expert down-proj -> bf16 slot outputs
  gemm_down_wmma<<<dim3(Hd / 128, Cd / 64, Ed), 256, 0, stream>>>(
      act, Wdbf, obuf, Md, Hd, /*store_bf16=*/1,
      (size_t)Cd * Md, (size_t)Hd * Md, (size_t)Cd * Hd);
  // 6) shared expert gate/up + SwiGLU
  gemm_swiglu_wmma<<<dim3(MSd / 128, Td / 64, 1), 256, 0, stream>>>(
      xbf, Sgbf, Subf, act_s, Hd, MSd, 0, 0, 0);
  // 7) shared expert down-proj straight into d_out (f32, overwrites all)
  gemm_down_wmma<<<dim3(Hd / 128, Td / 64, 1), 256, 0, stream>>>(
      act_s, Sdbf, out, MSd, Hd, /*store_bf16=*/0, 0, 0, 0);
  // 8) add weighted routed outputs
  combine_kernel<<<Td, 256, 0, stream>>>(obuf, topk_w, slot, out);
}
